// DeltaEGNN_13778255085802
// MI455X (gfx1250) — matrix operand order repaired and checked by reading, not compiled, not run on hardware
//
#include <hip/hip_runtime.h>

typedef _Float16 h16_t;
typedef _Float16 v16h __attribute__((ext_vector_type(16)));
typedef _Float16 v8h  __attribute__((ext_vector_type(8)));
typedef float    v8f  __attribute__((ext_vector_type(8)));

#define WPB 8   // waves per block (256 threads, wave32)

#define WAVE_SYNC() asm volatile("s_wait_dscnt 0" ::: "memory")

// silu via v_rcp_f32 (1 TRANS op) instead of the IEEE divide chain
static __device__ __forceinline__ float silu_f(float x) {
  return x * __builtin_amdgcn_rcpf(1.0f + __expf(-x));
}

static __device__ __forceinline__ v8f splat8(float v) {
  v8f r; for (int i = 0; i < 8; ++i) r[i] = v; return r;
}

static __device__ __forceinline__ v16h vzero16() {
  v16h r; for (int i = 0; i < 16; ++i) r[i] = (_Float16)0; return r;
}

// A-fragment load: lane (ko = 0 for lanes 0-15, 8 for lanes 16-31) holds
// row m: elems 0..7 = K ko..ko+7, elems 8..15 = K ko+16..ko+23 (16b A 16x32)
static __device__ __forceinline__ v16h load_arow(const h16_t* row, int ko) {
  const v8h* p0 = reinterpret_cast<const v8h*>(row + ko);
  const v8h* p1 = reinterpret_cast<const v8h*>(row + ko + 16);
  v8h lo = *p0, hi = *p1;
  v16h r;
  for (int t = 0; t < 8; ++t) { r[t] = lo[t]; r[8 + t] = hi[t]; }
  return r;
}

// B-fragment load from packed tile (32 lanes x 16 halves, contiguous per lane)
static __device__ __forceinline__ v16h load_btile(const h16_t* base, int lane) {
  const v8h* p = reinterpret_cast<const v8h*>(base + lane * 16);
  v8h lo = p[0], hi = p[1];
  v16h r;
  for (int t = 0; t < 8; ++t) { r[t] = lo[t]; r[8 + t] = hi[t]; }
  return r;
}

static __device__ __forceinline__ v8f wmma16(v16h a, v16h b, v8f c) {
  return __builtin_amdgcn_wmma_f32_16x16x32_f16(false, a, false, b, (short)0, c,
                                                false, false);
}

// ---------------------------------------------------------------------------
// Weight packing: B tile layout = lane n (0..15) holds col n, K kbase..kbase+7, kbase+16..kbase+23;
// lane n+16 holds col n, K kbase+8..kbase+15, kbase+24..kbase+31.  16 tiles per layer:
//  [0..5] msg1 (3 K-chunks x 2 N-tiles), [6..7] msg2, [8..9] coord1,
//  [10..13] node1 (2 K-chunks x 2 N-tiles), [14..15] node2
// ---------------------------------------------------------------------------
struct WPtrs { const float* W[15]; };  // per layer: msg1,msg2,coord1,node1,node2

__global__ void prep_kernel(WPtrs wp, h16_t* __restrict__ Bt) {
  int t = blockIdx.x * blockDim.x + threadIdx.x;
  int lane = t & 31;
  int tile = t >> 5;
  if (tile >= 48) return;
  int layer = tile / 16, which = tile % 16;
  const float* W; int k0, kvalid;
  if (which < 6)       { W = wp.W[layer * 5 + 0]; k0 = (which >> 1) * 32;        kvalid = 65; }
  else if (which < 8)  { W = wp.W[layer * 5 + 1]; k0 = 0;                        kvalid = 32; }
  else if (which < 10) { W = wp.W[layer * 5 + 2]; k0 = 0;                        kvalid = 32; }
  else if (which < 14) { W = wp.W[layer * 5 + 3]; k0 = ((which - 10) >> 1) * 32; kvalid = 64; }
  else                 { W = wp.W[layer * 5 + 4]; k0 = 0;                        kvalid = 32; }
  int n0 = (which & 1) * 16;
  int col = n0 + (lane & 15);
  int kbase = k0 + (lane >> 4) * 8;
  h16_t* out = Bt + (size_t)tile * 512 + lane * 16;
  for (int e = 0; e < 16; ++e) {
    int k = kbase + (e < 8 ? e : e + 8);
    float v = (k < kvalid) ? W[(size_t)k * 32 + col] : 0.0f;
    out[e] = (h16_t)v;
  }
}

// ---------------------------------------------------------------------------
__global__ void zero_kernel(float* __restrict__ p, size_t n) {
  size_t t = (size_t)blockIdx.x * blockDim.x + threadIdx.x;
  if (t < n) p[t] = 0.0f;
}

// ---------------------------------------------------------------------------
__global__ void embed_kernel(const float* __restrict__ x,
                             const float* __restrict__ pos_in,
                             const float* __restrict__ e0W, const float* __restrict__ e0b,
                             const float* __restrict__ e1W, const float* __restrict__ e1b,
                             float* __restrict__ h32, h16_t* __restrict__ h16,
                             float* __restrict__ pos_cur, float* __restrict__ wnode,
                             int N) {
  int n = blockIdx.x * blockDim.x + threadIdx.x;
  if (n >= N) return;
  float xin[25];
  for (int k = 0; k < 25; ++k) xin[k] = x[(size_t)n * 25 + k];
  float t[32];
  for (int o = 0; o < 32; ++o) {
    float acc = e0b[o];
    for (int k = 0; k < 25; ++k) acc += xin[k] * e0W[k * 32 + o];
    t[o] = silu_f(acc);
  }
  for (int o = 0; o < 32; ++o) {
    float acc = e1b[o];
    for (int k = 0; k < 32; ++k) acc += t[k] * e1W[k * 32 + o];
    h32[(size_t)n * 32 + o] = acc;
    h16[(size_t)n * 32 + o] = (h16_t)acc;
  }
  wnode[n] = fmaxf(xin[4], 1e-6f);
  for (int c = 0; c < 3; ++c) pos_cur[(size_t)n * 3 + c] = pos_in[(size_t)n * 3 + c];
}

// ---------------------------------------------------------------------------
// Edge kernel: one wave32 per 16-edge tile.  10 WMMAs per tile.
// Weight B-fragments staged once per block into LDS, shared by all 8 waves.
// ---------------------------------------------------------------------------
__global__ __launch_bounds__(256) void edge_kernel(
    const int* __restrict__ ei,          // [2*E]; j = ei[e], i = ei[E+e]
    const float* __restrict__ pos,
    const h16_t* __restrict__ h16,
    float* __restrict__ aggr_msg,
    float* __restrict__ aggr_coord,
    const h16_t* __restrict__ Bt,        // this layer's 16 packed tiles
    const float* __restrict__ m1b, const float* __restrict__ m2b,
    const float* __restrict__ c1b, const float* __restrict__ c2W,
    int E) {
  __shared__ h16_t lds_B[10 * 512];            // msg1(6), msg2(2), coord1(2)
  __shared__ h16_t lds_a[WPB][16 * 32];
  __shared__ float lds_m[WPB][16 * 32];
  __shared__ float lds_cw[WPB][16];

  // cooperative stage of weight fragments (all threads, before any exit)
  {
    const uint32_t* src = reinterpret_cast<const uint32_t*>(Bt);
    uint32_t* dst = reinterpret_cast<uint32_t*>(lds_B);
    for (int t = threadIdx.x; t < 10 * 512 / 2; t += 256) dst[t] = src[t];
  }
  __syncthreads();

  const int lane = threadIdx.x & 31;
  const int w    = threadIdx.x >> 5;
  const int tile = blockIdx.x * WPB + w;
  if (tile * 16 >= E) return;

  const int m    = lane & 15;
  const int hi   = lane >> 4;
  const int ko   = hi ? 8 : 0;
  const int ncol = m;

  bool ok = (tile * 16 + m) < E;
  int  e  = ok ? (tile * 16 + m) : 0;
  int  j  = ei[e];
  int  i  = ei[(size_t)E + e];

  float rx, ry, rz, d2;
  {
    float pix = pos[(size_t)i * 3 + 0], piy = pos[(size_t)i * 3 + 1], piz = pos[(size_t)i * 3 + 2];
    float pjx = pos[(size_t)j * 3 + 0], pjy = pos[(size_t)j * 3 + 1], pjz = pos[(size_t)j * 3 + 2];
    rx = pix - pjx; ry = piy - pjy; rz = piz - pjz;
    d2 = rx * rx + ry * ry + rz * rz;
  }
  if (!ok) d2 = 0.0f;

  // A fragments: chunk0 = h[i] (K 0..31), chunk1 = h[j] (K 32..63), chunk2 = d2 at K 64
  v16h a0 = load_arow(h16 + (size_t)i * 32, ko);
  v16h a1 = load_arow(h16 + (size_t)j * 32, ko);
  v16h a2 = vzero16();
  if (hi == 0) a2[0] = (_Float16)d2;

  // ---- msg1: 65 -> 32 -------------------------------------------------
  v8f c0 = splat8(m1b[ncol]);
  v8f c1 = splat8(m1b[ncol + 16]);
  c0 = wmma16(a0, load_btile(lds_B + 0 * 512, lane), c0);
  c1 = wmma16(a0, load_btile(lds_B + 1 * 512, lane), c1);
  c0 = wmma16(a1, load_btile(lds_B + 2 * 512, lane), c0);
  c1 = wmma16(a1, load_btile(lds_B + 3 * 512, lane), c1);
  c0 = wmma16(a2, load_btile(lds_B + 4 * 512, lane), c0);
  c1 = wmma16(a2, load_btile(lds_B + 5 * 512, lane), c1);
  // silu -> LDS (A-layout source)
  for (int r = 0; r < 8; ++r) {
    int m2 = r + 8 * hi;
    lds_a[w][m2 * 32 + ncol]      = (h16_t)silu_f(c0[r]);
    lds_a[w][m2 * 32 + ncol + 16] = (h16_t)silu_f(c1[r]);
  }
  WAVE_SYNC();
  v16h aA = load_arow(&lds_a[w][m * 32], ko);

  // ---- msg2: 32 -> 32 -------------------------------------------------
  v8f d0 = splat8(m2b[ncol]);
  v8f d1 = splat8(m2b[ncol + 16]);
  d0 = wmma16(aA, load_btile(lds_B + 6 * 512, lane), d0);
  d1 = wmma16(aA, load_btile(lds_B + 7 * 512, lane), d1);
  for (int r = 0; r < 8; ++r) {
    int m2 = r + 8 * hi;
    float v0 = silu_f(d0[r]), v1 = silu_f(d1[r]);
    lds_m[w][m2 * 32 + ncol]      = v0;
    lds_m[w][m2 * 32 + ncol + 16] = v1;
    lds_a[w][m2 * 32 + ncol]      = (h16_t)v0;
    lds_a[w][m2 * 32 + ncol + 16] = (h16_t)v1;
  }
  WAVE_SYNC();
  v16h aM = load_arow(&lds_a[w][m * 32], ko);

  // ---- coord1: 32 -> 32, coord2: 32 -> 1 ------------------------------
  v8f t0 = splat8(c1b[ncol]);
  v8f t1 = splat8(c1b[ncol + 16]);
  t0 = wmma16(aM, load_btile(lds_B + 8 * 512, lane), t0);
  t1 = wmma16(aM, load_btile(lds_B + 9 * 512, lane), t1);
  float wc0 = c2W[ncol], wc1 = c2W[ncol + 16];
  float p[8];
  for (int r = 0; r < 8; ++r)
    p[r] = silu_f(t0[r]) * wc0 + silu_f(t1[r]) * wc1;
  // reduce across the 16 lanes of each half (xor masks stay within half)
  for (int mask = 1; mask < 16; mask <<= 1)
    for (int r = 0; r < 8; ++r) p[r] += __shfl_xor(p[r], mask, 32);
  if (ncol == 0)
    for (int r = 0; r < 8; ++r) lds_cw[w][r + 8 * hi] = p[r];
  WAVE_SYNC();

  // ---- aggregation ----------------------------------------------------
  if (ok && hi == 0) {
    float cw = lds_cw[w][m];
    atomicAdd(&aggr_coord[(size_t)i * 3 + 0], rx * cw);
    atomicAdd(&aggr_coord[(size_t)i * 3 + 1], ry * cw);
    atomicAdd(&aggr_coord[(size_t)i * 3 + 2], rz * cw);
  }
  if (ok) {
    int cbase = hi * 16;
    float* dst = aggr_msg + (size_t)i * 32 + cbase;
    const float* src = &lds_m[w][m * 32 + cbase];
    for (int t = 0; t < 16; ++t) atomicAdd(&dst[t], src[t]);
  }
}

// ---------------------------------------------------------------------------
// Node kernel: one wave32 per 16-node tile.  6 WMMAs per tile.
// ---------------------------------------------------------------------------
__global__ __launch_bounds__(256) void node_kernel(
    float* __restrict__ h32, h16_t* __restrict__ h16,
    float* __restrict__ pos,
    const float* __restrict__ aggr_msg, const float* __restrict__ aggr_coord,
    const h16_t* __restrict__ Bt,        // full 16-tile layer base
    const float* __restrict__ n1b, const float* __restrict__ n2b,
    int N) {
  __shared__ h16_t lds_B[6 * 512];             // node1(4), node2(2)
  __shared__ h16_t lds_a[WPB][16 * 32];

  {
    const uint32_t* src = reinterpret_cast<const uint32_t*>(Bt + 10 * 512);
    uint32_t* dst = reinterpret_cast<uint32_t*>(lds_B);
    for (int t = threadIdx.x; t < 6 * 512 / 2; t += 256) dst[t] = src[t];
  }
  __syncthreads();

  const int lane = threadIdx.x & 31;
  const int w    = threadIdx.x >> 5;
  const int tile = blockIdx.x * WPB + w;
  if (tile * 16 >= N) return;

  const int m    = lane & 15;
  const int hi   = lane >> 4;
  const int ko   = hi ? 8 : 0;
  const int ncol = m;

  bool ok_row = (tile * 16 + m) < N;
  int  nd     = ok_row ? (tile * 16 + m) : 0;

  v16h a0 = load_arow(h16 + (size_t)nd * 32, ko);
  v16h a1;
  {
    const float* ar = aggr_msg + (size_t)nd * 32;
    for (int t = 0; t < 8; ++t) {
      a1[t]     = (_Float16)ar[ko + t];
      a1[8 + t] = (_Float16)ar[ko + 16 + t];
    }
  }
  // node1: [h | aggr_msg] (64) -> 32
  v8f c0 = splat8(n1b[ncol]);
  v8f c1 = splat8(n1b[ncol + 16]);
  c0 = wmma16(a0, load_btile(lds_B + 0 * 512, lane), c0);
  c1 = wmma16(a0, load_btile(lds_B + 1 * 512, lane), c1);
  c0 = wmma16(a1, load_btile(lds_B + 2 * 512, lane), c0);
  c1 = wmma16(a1, load_btile(lds_B + 3 * 512, lane), c1);
  for (int r = 0; r < 8; ++r) {
    int m2 = r + 8 * hi;
    lds_a[w][m2 * 32 + ncol]      = (h16_t)silu_f(c0[r]);
    lds_a[w][m2 * 32 + ncol + 16] = (h16_t)silu_f(c1[r]);
  }
  WAVE_SYNC();
  v16h aU = load_arow(&lds_a[w][m * 32], ko);

  // node2: 32 -> 32
  v8f d0 = splat8(n2b[ncol]);
  v8f d1 = splat8(n2b[ncol + 16]);
  d0 = wmma16(aU, load_btile(lds_B + 4 * 512, lane), d0);
  d1 = wmma16(aU, load_btile(lds_B + 5 * 512, lane), d1);

  // residual h += upd  (C-layout scatter)
  for (int r = 0; r < 8; ++r) {
    int m2 = r + 8 * hi;
    int node2 = tile * 16 + m2;
    if (node2 < N) {
      size_t o0 = (size_t)node2 * 32 + ncol;
      float nv0 = h32[o0] + d0[r];
      float nv1 = h32[o0 + 16] + d1[r];
      h32[o0] = nv0;            h32[o0 + 16] = nv1;
      h16[o0] = (h16_t)nv0;     h16[o0 + 16] = (h16_t)nv1;
    }
  }
  if (hi == 0 && ok_row) {
    for (int c = 0; c < 3; ++c)
      pos[(size_t)nd * 3 + c] += aggr_coord[(size_t)nd * 3 + c];
  }
}

// ---------------------------------------------------------------------------
__global__ void pool_kernel(const float* __restrict__ h32,
                            const float* __restrict__ wnode,
                            const int* __restrict__ batch,
                            float* __restrict__ sw_h, float* __restrict__ sw,
                            int N) {
  size_t t = (size_t)blockIdx.x * blockDim.x + threadIdx.x;
  if (t >= (size_t)N * 32) return;
  int n = (int)(t >> 5), c = (int)(t & 31);
  int b = batch[n];
  float wv = wnode[n];
  atomicAdd(&sw_h[(size_t)b * 32 + c], h32[t] * wv);
  if (c == 0) atomicAdd(&sw[b], wv);
}

__global__ void head_kernel(const float* __restrict__ ga,
                            const float* __restrict__ sw_h, const float* __restrict__ sw,
                            const float* __restrict__ g0W, const float* __restrict__ g0b,
                            const float* __restrict__ g1W, const float* __restrict__ g1b,
                            const float* __restrict__ h0W, const float* __restrict__ h0b,
                            const float* __restrict__ h1W, const float* __restrict__ h1b,
                            float* __restrict__ out, int G) {
  int g = blockIdx.x * blockDim.x + threadIdx.x;
  if (g >= G) return;
  float t16[16];
  for (int o = 0; o < 16; ++o) {
    float acc = g0b[o];
    for (int k = 0; k < 3; ++k) acc += ga[(size_t)g * 3 + k] * g0W[k * 16 + o];
    t16[o] = silu_f(acc);
  }
  float ge[16];
  for (int o = 0; o < 16; ++o) {
    float acc = g1b[o];
    for (int k = 0; k < 16; ++k) acc += t16[k] * g1W[k * 16 + o];
    ge[o] = acc;
  }
  float inv = __builtin_amdgcn_rcpf(sw[g]);
  float comb[48];
  for (int k = 0; k < 32; ++k) comb[k] = sw_h[(size_t)g * 32 + k] * inv;
  for (int k = 0; k < 16; ++k) comb[32 + k] = ge[k];
  float acc2 = h1b[0];
  for (int o = 0; o < 32; ++o) {
    float acc = h0b[o];
    for (int k = 0; k < 48; ++k) acc += comb[k] * h0W[k * 32 + o];
    acc2 += silu_f(acc) * h1W[o];
  }
  out[g] = acc2;
}

// ---------------------------------------------------------------------------
extern "C" void kernel_launch(void* const* d_in, const int* in_sizes, int n_in,
                              void* d_out, int out_size, void* d_ws, size_t ws_size,
                              hipStream_t stream) {
  const float* x    = (const float*)d_in[0];
  const float* pos  = (const float*)d_in[1];
  const int*   batch= (const int*)d_in[2];
  const int*   eidx = (const int*)d_in[3];
  const float* ga   = (const float*)d_in[4];
  const int N = in_sizes[0] / 25;
  const int E = in_sizes[3] / 2;
  const int G = in_sizes[4] / 3;

  // params leaves (jax pytree order: convs, embed, genc, head; dicts sorted)
  const float *c1W[3], *c1b[3], *c2W[3], *m1W[3], *m1b[3], *m2W[3], *m2b[3];
  const float *n1W[3], *n1b[3], *n2W[3], *n2b[3];
  for (int L = 0; L < 3; ++L) {
    const int b = 5 + 11 * L;
    c1W[L] = (const float*)d_in[b + 0];  c1b[L] = (const float*)d_in[b + 1];
    c2W[L] = (const float*)d_in[b + 2];
    m1W[L] = (const float*)d_in[b + 3];  m1b[L] = (const float*)d_in[b + 4];
    m2W[L] = (const float*)d_in[b + 5];  m2b[L] = (const float*)d_in[b + 6];
    n1W[L] = (const float*)d_in[b + 7];  n1b[L] = (const float*)d_in[b + 8];
    n2W[L] = (const float*)d_in[b + 9];  n2b[L] = (const float*)d_in[b + 10];
  }
  const int pb = 5 + 33;
  const float* e0W = (const float*)d_in[pb + 0];
  const float* e0b = (const float*)d_in[pb + 1];
  const float* e1W = (const float*)d_in[pb + 2];
  const float* e1b = (const float*)d_in[pb + 3];
  const float* g0W = (const float*)d_in[pb + 4];
  const float* g0b = (const float*)d_in[pb + 5];
  const float* g1W = (const float*)d_in[pb + 6];
  const float* g1b = (const float*)d_in[pb + 7];
  const float* h0W = (const float*)d_in[pb + 8];
  const float* h0b = (const float*)d_in[pb + 9];
  const float* h1W = (const float*)d_in[pb + 10];
  const float* h1b = (const float*)d_in[pb + 11];

  // workspace layout
  char* ws = (char*)d_ws;
  size_t off = 0;
  auto AL = [](size_t o) { return (o + 255) & ~(size_t)255; };
  float* h32       = (float*)(ws + off); off = AL(off + (size_t)N * 32 * 4);
  h16_t* h16p      = (h16_t*)(ws + off); off = AL(off + (size_t)N * 32 * 2);
  float* pos_cur   = (float*)(ws + off); off = AL(off + (size_t)N * 3 * 4);
  float* aggr_msg  = (float*)(ws + off); off = AL(off + (size_t)N * 32 * 4);
  float* aggr_crd  = (float*)(ws + off); off = AL(off + (size_t)N * 3 * 4);
  float* wnode     = (float*)(ws + off); off = AL(off + (size_t)N * 4);
  h16_t* packedB   = (h16_t*)(ws + off); off = AL(off + (size_t)3 * 16 * 512 * 2);
  float* sw_h      = (float*)(ws + off); off = AL(off + (size_t)G * 32 * 4);
  float* swp       = (float*)(ws + off); off = AL(off + (size_t)G * 4);

  // 1. pack WMMA B tiles
  WPtrs wp;
  for (int L = 0; L < 3; ++L) {
    wp.W[L * 5 + 0] = m1W[L];
    wp.W[L * 5 + 1] = m2W[L];
    wp.W[L * 5 + 2] = c1W[L];
    wp.W[L * 5 + 3] = n1W[L];
    wp.W[L * 5 + 4] = n2W[L];
  }
  prep_kernel<<<6, 256, 0, stream>>>(wp, packedB);

  // 2. embed
  embed_kernel<<<(N + 255) / 256, 256, 0, stream>>>(
      x, pos, e0W, e0b, e1W, e1b, h32, h16p, pos_cur, wnode, N);

  // 3. conv layers
  const int etiles = (E + 15) / 16;
  const int eblk   = (etiles + WPB - 1) / WPB;
  const int ntiles = (N + 15) / 16;
  const int nblk   = (ntiles + WPB - 1) / WPB;
  for (int L = 0; L < 3; ++L) {
    zero_kernel<<<((size_t)N * 32 + 255) / 256, 256, 0, stream>>>(aggr_msg, (size_t)N * 32);
    zero_kernel<<<((size_t)N * 3 + 255) / 256, 256, 0, stream>>>(aggr_crd, (size_t)N * 3);
    const h16_t* Bt = packedB + (size_t)L * 16 * 512;
    edge_kernel<<<eblk, 256, 0, stream>>>(eidx, pos_cur, h16p, aggr_msg, aggr_crd,
                                          Bt, m1b[L], m2b[L], c1b[L], c2W[L], E);
    node_kernel<<<nblk, 256, 0, stream>>>(h32, h16p, pos_cur, aggr_msg, aggr_crd,
                                          Bt, n1b[L], n2b[L], N);
  }

  // 4. pooling + head
  zero_kernel<<<((size_t)G * 32 + 255) / 256, 256, 0, stream>>>(sw_h, (size_t)G * 32);
  zero_kernel<<<((size_t)G + 255) / 256, 256, 0, stream>>>(swp, (size_t)G);
  pool_kernel<<<((size_t)N * 32 + 255) / 256, 256, 0, stream>>>(h32, wnode, batch,
                                                                sw_h, swp, N);
  head_kernel<<<(G + 63) / 64, 64, 0, stream>>>(ga, sw_h, swp, g0W, g0b, g1W, g1b,
                                                h0W, h0b, h1W, h1b, (float*)d_out, G);
}
